// Attention_9208409882893
// MI455X (gfx1250) — matrix operand order repaired and checked by reading, not compiled, not run on hardware
//
#include <hip/hip_runtime.h>
#include <hip/hip_bf16.h>
#include <math.h>

// Problem constants: B=32, Q=K=512, C=256, H=8, DH=32, OUT=256
typedef __attribute__((ext_vector_type(16))) __bf16 v16bf;
typedef __attribute__((ext_vector_type(8)))  __bf16 v8bf;
typedef __attribute__((ext_vector_type(8)))  float  v8f;

union Frag16 { v16bf v; v8bf h[2]; };

__device__ __forceinline__ v8f vzero8() {
    v8f z;
#pragma unroll
    for (int i = 0; i < 8; ++i) z[i] = 0.0f;
    return z;
}

__device__ __forceinline__ __bf16 f2bf(float x) { return (__bf16)x; }

__device__ __forceinline__ float rmax16(float t) {
#pragma unroll
    for (int off = 1; off < 16; off <<= 1)
        t = fmaxf(t, __shfl_xor(t, off, 32));
    return t;
}

// ---------------------------------------------------------------------------
// Generic 256-K GEMM: D = act(A[M,256] @ W[256,256]) with per-mode epilogue.
// MODE 0: q  -> bf16 [B][H][512][32], *scale
// MODE 1: k  -> bf16 [B][H][512][32]
// MODE 2: vT -> bf16 [B][H][32][512]
// MODE 3: gate = sigmoid(x + gating_b) -> f32 [M][256]
// MODE 4: out = x + output_b -> f32 [M][256]
// Block = 128 threads (4 waves); block tile 64x64; wave tile 16x64.
// ---------------------------------------------------------------------------
template <int MODE>
__global__ __launch_bounds__(128)
void gemm256(const float* __restrict__ A, const float* __restrict__ W,
             const float* __restrict__ bvec, void* __restrict__ outp, float scale)
{
    __shared__ __bf16 As[64][32];   // [m][k]
    __shared__ __bf16 Ws[64][32];   // [n][k] (transposed stage)
    const int tid  = threadIdx.x;
    const int lane = tid & 31;
    const int wave = tid >> 5;
    const int tileM = blockIdx.y * 64;
    const int tileN = blockIdx.x * 64;

    v8f acc[4];
#pragma unroll
    for (int i = 0; i < 4; ++i) acc[i] = vzero8();

    const int mloc = lane & 15;
    const int ksel = (lane < 16) ? 0 : 8;    // A-frag K base (per ISA layout)
    const int kb16 = (lane < 16) ? 0 : 16;   // unused (B frag also uses ksel)

    for (int k0 = 0; k0 < 256; k0 += 32) {
        // Stage A tile (64x32 f32 -> bf16) and W tile (32x64, transposed)
#pragma unroll
        for (int i = 0; i < 4; ++i) {
            int qd  = tid + i * 128;         // 0..511 float4 slots
            int row = qd >> 3;               // A: 8 float4 per row of 32
            int kq  = (qd & 7) << 2;
            const float4 av = *reinterpret_cast<const float4*>(
                A + (size_t)(tileM + row) * 256 + k0 + kq);
            As[row][kq + 0] = f2bf(av.x);
            As[row][kq + 1] = f2bf(av.y);
            As[row][kq + 2] = f2bf(av.z);
            As[row][kq + 3] = f2bf(av.w);

            int krow = qd >> 4;              // W: 16 float4 per row of 64
            int nq   = (qd & 15) << 2;
            const float4 wv = *reinterpret_cast<const float4*>(
                W + (size_t)(k0 + krow) * 256 + tileN + nq);
            Ws[nq + 0][krow] = f2bf(wv.x);
            Ws[nq + 1][krow] = f2bf(wv.y);
            Ws[nq + 2][krow] = f2bf(wv.z);
            Ws[nq + 3][krow] = f2bf(wv.w);
        }
        __syncthreads();

        Frag16 af;
        af.h[0] = *reinterpret_cast<const v8bf*>(&As[wave * 16 + mloc][ksel]);
        af.h[1] = *reinterpret_cast<const v8bf*>(&As[wave * 16 + mloc][16 + ksel]);
#pragma unroll
        for (int nb = 0; nb < 4; ++nb) {
            Frag16 bfr;
            bfr.h[0] = *reinterpret_cast<const v8bf*>(&Ws[nb * 16 + mloc][ksel]);
            bfr.h[1] = *reinterpret_cast<const v8bf*>(&Ws[nb * 16 + mloc][16 + ksel]);
            acc[nb] = __builtin_amdgcn_wmma_f32_16x16x32_bf16(
                false, af.v, false, bfr.v, (short)0, acc[nb], false, false);
        }
        __syncthreads();
    }

    const int halfsel = lane >> 4;
#pragma unroll
    for (int nb = 0; nb < 4; ++nb) {
#pragma unroll
        for (int r = 0; r < 8; ++r) {
            int gm = tileM + wave * 16 + r + halfsel * 8;  // C/D row mapping
            int gn = tileN + nb * 16 + mloc;               // C/D col mapping
            float val = acc[nb][r];
            if (MODE == 0 || MODE == 1) {
                val *= scale;
                int b = gm >> 9, row = gm & 511;
                int h = gn >> 5, d = gn & 31;
                ((__bf16*)outp)[(((size_t)b * 8 + h) * 512 + row) * 32 + d] = f2bf(val);
            } else if (MODE == 2) {
                int b = gm >> 9, row = gm & 511;
                int h = gn >> 5, d = gn & 31;
                ((__bf16*)outp)[(((size_t)b * 8 + h) * 32 + d) * 512 + row] = f2bf(val);
            } else if (MODE == 3) {
                float g = val + bvec[gn];
                g = 1.0f / (1.0f + __expf(-g));
                ((float*)outp)[(size_t)gm * 256 + gn] = g;
            } else {
                ((float*)outp)[(size_t)gm * 256 + gn] = val + bvec[gn];
            }
        }
    }
}

// ---------------------------------------------------------------------------
// Flash attention v2: one wave per (b, h, 16-row q tile). DH=32 == WMMA K-dim.
// 64-key softmax pass (4 logit tiles per online update); row-sums computed by
// WMMA against an all-ones B matrix (replaces shuffle-add reductions); P is
// converted D-layout -> A-layout through a per-wave LDS buffer.
// ---------------------------------------------------------------------------
__global__ __launch_bounds__(128)
void attn_kernel(const __bf16* __restrict__ q, const __bf16* __restrict__ k,
                 const __bf16* __restrict__ vT, const float* __restrict__ bias,
                 const float* __restrict__ nbias, const float* __restrict__ gate,
                 float* __restrict__ wa)
{
    __shared__ __bf16 Plds[4][16][64];
    const int tid  = threadIdx.x;
    const int lane = tid & 31;
    const int wave = tid >> 5;
    const int task = blockIdx.x * 4 + wave;  // 8192 = 32*8*32 tasks
    const int qt = task & 31;
    const int h  = (task >> 5) & 7;
    const int b  = task >> 8;
    const int qBase = qt * 16;

    const __bf16* qp = q  + (((size_t)b * 8 + h) * 512 + qBase) * 32;
    const __bf16* kp = k  + ((size_t)b * 8 + h) * 512 * 32;
    const __bf16* vp = vT + ((size_t)b * 8 + h) * 32 * 512;
    const float* biasp = bias  + ((size_t)b * 512 + qBase) * 512;
    const float* nbp   = nbias + ((size_t)h * 512 + qBase) * 512;

    const int mloc  = lane & 15;
    const int ksel  = (lane < 16) ? 0 : 8;
    const int kb16  = (lane < 16) ? 0 : 16;
    const int half8 = (lane >> 4) * 8;

    Frag16 aq;  // q tile 16x32, reused for all key tiles
    aq.h[0] = *reinterpret_cast<const v8bf*>(qp + (size_t)mloc * 32 + ksel);
    aq.h[1] = *reinterpret_cast<const v8bf*>(qp + (size_t)mloc * 32 + 16 + ksel);

    Frag16 ones;  // all-ones B matrix for WMMA row-sums
#pragma unroll
    for (int i = 0; i < 16; ++i) ones.v[i] = (__bf16)1.0f;

    v8f O0 = vzero8(), O1 = vzero8();
    float mrow[8], lrow[8], alpha[8];
#pragma unroll
    for (int r = 0; r < 8; ++r) { mrow[r] = -3.0e38f; lrow[r] = 0.0f; }

    for (int kk = 0; kk < 512; kk += 64) {
        // ---- logits: 4 tiles of 16 keys -------------------------------
        v8f S[4];
#pragma unroll
        for (int j = 0; j < 4; ++j) {
            Frag16 bk;
            const __bf16* kjp = kp + (size_t)(kk + j * 16 + mloc) * 32 + ksel;
            bk.h[0] = *reinterpret_cast<const v8bf*>(kjp);
            bk.h[1] = *reinterpret_cast<const v8bf*>(kjp + 16);
            S[j] = __builtin_amdgcn_wmma_f32_16x16x32_bf16(
                false, aq.v, false, bk.v, (short)0, vzero8(), false, false);
        }

        // ---- online softmax over the 64-key strip ---------------------
#pragma unroll
        for (int r = 0; r < 8; ++r) {
            const int qr = r + half8;
            const size_t bi = (size_t)qr * 512 + kk + mloc;
            float s0 = S[0][r] + biasp[bi]      + nbp[bi];
            float s1 = S[1][r] + biasp[bi + 16] + nbp[bi + 16];
            float s2 = S[2][r] + biasp[bi + 32] + nbp[bi + 32];
            float s3 = S[3][r] + biasp[bi + 48] + nbp[bi + 48];
            s0 = fminf(fmaxf(s0, -1.0e8f), 1.0e8f);
            s1 = fminf(fmaxf(s1, -1.0e8f), 1.0e8f);
            s2 = fminf(fmaxf(s2, -1.0e8f), 1.0e8f);
            s3 = fminf(fmaxf(s3, -1.0e8f), 1.0e8f);
            float tmax = rmax16(fmaxf(fmaxf(s0, s1), fmaxf(s2, s3)));
            float mn = fmaxf(mrow[r], tmax);
            alpha[r] = __expf(mrow[r] - mn);
            mrow[r]  = mn;
            Plds[wave][qr][mloc]      = f2bf(__expf(s0 - mn));
            Plds[wave][qr][16 + mloc] = f2bf(__expf(s1 - mn));
            Plds[wave][qr][32 + mloc] = f2bf(__expf(s2 - mn));
            Plds[wave][qr][48 + mloc] = f2bf(__expf(s3 - mn));
            O0[r] *= alpha[r];
            O1[r] *= alpha[r];
        }

        // ---- P: D-layout -> A-layout via LDS (in-wave dscnt ordering) -
        Frag16 p0, p1;
        p0.h[0] = *reinterpret_cast<const v8bf*>(&Plds[wave][mloc][ksel]);
        p0.h[1] = *reinterpret_cast<const v8bf*>(&Plds[wave][mloc][16 + ksel]);
        p1.h[0] = *reinterpret_cast<const v8bf*>(&Plds[wave][mloc][32 + ksel]);
        p1.h[1] = *reinterpret_cast<const v8bf*>(&Plds[wave][mloc][48 + ksel]);

        // ---- row sums via WMMA against all-ones B ---------------------
        v8f rs = __builtin_amdgcn_wmma_f32_16x16x32_bf16(
            false, p0.v, false, ones.v, (short)0, vzero8(), false, false);
        rs = __builtin_amdgcn_wmma_f32_16x16x32_bf16(
            false, p1.v, false, ones.v, (short)0, rs, false, false);
#pragma unroll
        for (int r = 0; r < 8; ++r)
            lrow[r] = lrow[r] * alpha[r] + rs[r];

        // ---- P @ V accumulation ---------------------------------------
#pragma unroll
        for (int t = 0; t < 2; ++t) {     // d tile: cols t*16..t*16+15
            Frag16 bva, bvb;
            const __bf16* va = vp + (size_t)(t * 16 + mloc) * 512 + kk + ksel;
            const __bf16* vb = va + 32;
            bva.h[0] = *reinterpret_cast<const v8bf*>(va);
            bva.h[1] = *reinterpret_cast<const v8bf*>(va + 16);
            bvb.h[0] = *reinterpret_cast<const v8bf*>(vb);
            bvb.h[1] = *reinterpret_cast<const v8bf*>(vb + 16);
            v8f& Ot = (t == 0) ? O0 : O1;
            Ot = __builtin_amdgcn_wmma_f32_16x16x32_bf16(
                false, p0.v, false, bva.v, (short)0, Ot, false, false);
            Ot = __builtin_amdgcn_wmma_f32_16x16x32_bf16(
                false, p1.v, false, bvb.v, (short)0, Ot, false, false);
        }
    }

#pragma unroll
    for (int r = 0; r < 8; ++r) {
        const int gq = qBase + r + half8;
        const float inv = 1.0f / lrow[r];
        const size_t grow  = (size_t)b * 512 + gq;
        const size_t gbase = (grow * 8 + h) * 32;
        float o0 = O0[r] * inv * gate[gbase + mloc];
        float o1 = O1[r] * inv * gate[gbase + 16 + mloc];
        wa[grow * 256 + h * 32 + mloc]      = o0;
        wa[grow * 256 + h * 32 + 16 + mloc] = o1;
    }
}

// ---------------------------------------------------------------------------
extern "C" void kernel_launch(void* const* d_in, const int* in_sizes, int n_in,
                              void* d_out, int out_size, void* d_ws, size_t ws_size,
                              hipStream_t stream)
{
    const float* q_data   = (const float*)d_in[0];
    const float* m_data   = (const float*)d_in[1];
    const float* bias     = (const float*)d_in[2];
    const float* nbias    = (const float*)d_in[3];
    const float* query_w  = (const float*)d_in[4];
    const float* key_w    = (const float*)d_in[5];
    const float* value_w  = (const float*)d_in[6];
    const float* gating_w = (const float*)d_in[7];
    const float* gating_b = (const float*)d_in[8];
    const float* output_w = (const float*)d_in[9];
    const float* output_b = (const float*)d_in[10];

    // Workspace layout (bytes): q 8MiB | k 8MiB | vT 8MiB | gate 16MiB | wa 16MiB
    char* ws = (char*)d_ws;
    __bf16* q_ws    = (__bf16*)(ws);
    __bf16* k_ws    = (__bf16*)(ws + 8388608);
    __bf16* vT_ws   = (__bf16*)(ws + 16777216);
    float*  gate_ws = (float*) (ws + 25165824);
    float*  wa_ws   = (float*) (ws + 41943040);

    dim3 gg(4, 256), bb(128);
    const float qscale = 0.17677669529663687f;  // 1/sqrt(DH=32)
    gemm256<0><<<gg, bb, 0, stream>>>(q_data, query_w,  nullptr,  (void*)q_ws,    qscale);
    gemm256<1><<<gg, bb, 0, stream>>>(m_data, key_w,    nullptr,  (void*)k_ws,    1.0f);
    gemm256<2><<<gg, bb, 0, stream>>>(m_data, value_w,  nullptr,  (void*)vT_ws,   1.0f);
    gemm256<3><<<gg, bb, 0, stream>>>(q_data, gating_w, gating_b, (void*)gate_ws, 1.0f);

    attn_kernel<<<2048, 128, 0, stream>>>(q_ws, k_ws, vT_ws, bias, nbias, gate_ws, wa_ws);

    gemm256<4><<<gg, bb, 0, stream>>>(wa_ws, output_w, output_b, d_out, 1.0f);
}
